// QuantizedAntPolicy_23888608100973
// MI455X (gfx1250) — 8-bit matrix operand order repaired and checked by reading, compiled, not run on hardware
//
#include <hip/hip_runtime.h>
#include <hip/hip_bf16.h>
#include <stdint.h>


typedef int v8i __attribute__((ext_vector_type(8)));

__device__ __forceinline__ float clampf(float v, float lo, float hi) {
    return fminf(fmaxf(v, lo), hi);
}

__global__ void k_init(unsigned* s1max_bits, unsigned* hmax_bits) {
    if (threadIdx.x == 0) { *s1max_bits = 0u; *hmax_bits = 0u; }
}

__global__ __launch_bounds__(256) void k_absmax(const float* __restrict__ x,
                                                int n, unsigned* out_bits) {
    __shared__ float red[256];
    float m = 0.0f;
    for (int i = blockIdx.x * blockDim.x + threadIdx.x; i < n;
         i += gridDim.x * blockDim.x)
        m = fmaxf(m, fabsf(x[i]));
    red[threadIdx.x] = m;
    __syncthreads();
    for (int s = 128; s > 0; s >>= 1) {
        if (threadIdx.x < s) red[threadIdx.x] = fmaxf(red[threadIdx.x], red[threadIdx.x + s]);
        __syncthreads();
    }
    if (threadIdx.x == 0) atomicMax(out_bits, __float_as_uint(red[0]));
}

__global__ __launch_bounds__(256) void k_quant_x(const float* __restrict__ x,
                                                 const unsigned* __restrict__ s1max_bits,
                                                 int8_t* __restrict__ xq, int n4) {
    int i = blockIdx.x * blockDim.x + threadIdx.x;
    if (i >= n4) return;
    const float inv = 127.0f / __uint_as_float(*s1max_bits);
    float4 v = reinterpret_cast<const float4*>(x)[i];
    char4 q;
    q.x = (signed char)clampf(rintf(v.x * inv), -128.0f, 127.0f);
    q.y = (signed char)clampf(rintf(v.y * inv), -128.0f, 127.0f);
    q.z = (signed char)clampf(rintf(v.z * inv), -128.0f, 127.0f);
    q.w = (signed char)clampf(rintf(v.w * inv), -128.0f, 127.0f);
    reinterpret_cast<char4*>(xq)[i] = q;
}

__global__ __launch_bounds__(256) void k_quant_w(const float* __restrict__ W,
                                                 const float* __restrict__ b, int K,
                                                 const unsigned* __restrict__ act_bits,
                                                 float act_div,
                                                 int8_t* __restrict__ Wq,
                                                 int* __restrict__ b_int,
                                                 float* __restrict__ fs_out,
                                                 float* __restrict__ fs_mirror,
                                                 float* __restrict__ s_out) {
    const int o = blockIdx.x;
    __shared__ float red[256];
    float m = 0.0f;
    for (int k = threadIdx.x; k < K; k += blockDim.x)
        m = fmaxf(m, fabsf(W[(size_t)o * K + k]));
    red[threadIdx.x] = m;
    __syncthreads();
    for (int s = 128; s > 0; s >>= 1) {
        if (threadIdx.x < s) red[threadIdx.x] = fmaxf(red[threadIdx.x], red[threadIdx.x + s]);
        __syncthreads();
    }
    const float ws  = red[0] / 127.0f;
    const float inv = 1.0f / ws;
    for (int k = threadIdx.x; k < K; k += blockDim.x) {
        float q = rintf(W[(size_t)o * K + k] * inv);
        Wq[(size_t)o * K + k] = (int8_t)clampf(q, -128.0f, 127.0f);
    }
    if (threadIdx.x == 0) {
        const float act_scale = __uint_as_float(*act_bits) / act_div;
        const float f = act_scale * ws;
        fs_out[o] = f;
        b_int[o]  = (int)rintf(b[o] / f);
        if (fs_mirror) fs_mirror[o] = f;
        if (s_out && o == 0) *s_out = act_scale;
    }
}

template <int K, bool A_SIGNED, int MODE>
__global__ __launch_bounds__(256) void gemm_i8(const int8_t* __restrict__ A,
                                               const int8_t* __restrict__ Bw,
                                               const int* __restrict__ bias_int,
                                               const float* __restrict__ fs,
                                               const unsigned* __restrict__ hmax_bits,
                                               float* __restrict__ outF,
                                               unsigned char* __restrict__ outU8,
                                               unsigned* __restrict__ gmax,
                                               int N) {
    constexpr int BM = 128, BN = 64, BK = 64;
    constexpr int LDA = 72, LDB = 72;
    __shared__ __align__(16) unsigned char ldsA[BM * LDA];
    __shared__ __align__(16) unsigned char ldsB[BN * LDB];
    __shared__ float red[256];

    const int t    = threadIdx.x;
    const int lane = t & 31;
    const int wave = t >> 5;
    const int wm   = (wave & 3) * 32;
    const int wn   = (wave >> 2) * 32;
    const int half = lane >> 4;
    const int l16  = lane & 15;
    const int m_blk = blockIdx.y * BM;
    const int n_blk = blockIdx.x * BN;

    v8i acc[2][2] = {};

    for (int k0 = 0; k0 < K; k0 += BK) {
        unsigned long long ar[4], br[2];
#pragma unroll
        for (int c = 0; c < 4; ++c) {
            int ch = t + c * 256, r = ch >> 3, col = (ch & 7) << 3;
            ar[c] = *reinterpret_cast<const unsigned long long*>(
                A + (size_t)(m_blk + r) * K + k0 + col);
        }
#pragma unroll
        for (int c = 0; c < 2; ++c) {
            int ch = t + c * 256, r = ch >> 3, col = (ch & 7) << 3;
            br[c] = *reinterpret_cast<const unsigned long long*>(
                Bw + (size_t)(n_blk + r) * K + k0 + col);
        }
        __syncthreads();
#pragma unroll
        for (int c = 0; c < 4; ++c) {
            int ch = t + c * 256, r = ch >> 3, col = (ch & 7) << 3;
            *reinterpret_cast<unsigned long long*>(ldsA + r * LDA + col) = ar[c];
        }
#pragma unroll
        for (int c = 0; c < 2; ++c) {
            int ch = t + c * 256, r = ch >> 3, col = (ch & 7) << 3;
            *reinterpret_cast<unsigned long long*>(ldsB + r * LDB + col) = br[c];
        }
        __syncthreads();

        v8i af[2], bf[2];
#pragma unroll
        for (int i = 0; i < 2; ++i) {
            const unsigned char* base = ldsA + (wm + i * 16 + l16) * LDA + half * 8;
#pragma unroll
            for (int v = 0; v < 8; ++v) {
                int koff = ((v >> 1) << 4) + ((v & 1) << 2);
                af[i][v] = *reinterpret_cast<const int*>(base + koff);
            }
        }
#pragma unroll
        for (int j = 0; j < 2; ++j) {
            const unsigned char* base = ldsB + (wn + j * 16 + l16) * LDB + half * 8;
#pragma unroll
            for (int v = 0; v < 8; ++v) {
                int koff = ((v >> 1) << 4) + ((v & 1) << 2);
                bf[j][v] = *reinterpret_cast<const int*>(base + koff);
            }
        }
#pragma unroll
        for (int i = 0; i < 2; ++i)
#pragma unroll
            for (int j = 0; j < 2; ++j)
                acc[i][j] = __builtin_amdgcn_wmma_i32_16x16x64_iu8(
                    A_SIGNED, af[i], true, bf[j], acc[i][j], false, false);
        __syncthreads();
    }

    float inv_s2 = 0.0f;
    if (MODE == 1) inv_s2 = 255.0f / __uint_as_float(*hmax_bits);
    float lmax = 0.0f;
#pragma unroll
    for (int i = 0; i < 2; ++i) {
#pragma unroll
        for (int j = 0; j < 2; ++j) {
            const int n   = n_blk + wn + j * 16 + l16;
            const float f = fs[n];
            const int  bi = bias_int[n];
            const int  m0 = m_blk + wm + i * 16 + half * 8;
#pragma unroll
            for (int r = 0; r < 8; ++r) {
                const float h = (float)(acc[i][j][r] + bi) * f;
                if (MODE == 0) {
                    lmax = fmaxf(lmax, h);
                } else if (MODE == 1) {
                    float q = clampf(rintf(h * inv_s2), 0.0f, 255.0f);
                    outU8[(size_t)(m0 + r) * N + n] = (unsigned char)q;
                } else {
                    outF[(size_t)(m0 + r) * N + n] = fmaxf(h, 0.0f);
                }
            }
        }
    }
    if (MODE == 0) {
        red[t] = lmax;
        __syncthreads();
        for (int s = 128; s > 0; s >>= 1) {
            if (t < s) red[t] = fmaxf(red[t], red[t + s]);
            __syncthreads();
        }
        if (t == 0) atomicMax(gmax, __float_as_uint(red[0]));
    }
}

extern "C" void kernel_launch(void* const* d_in, const int* in_sizes, int n_in,
                              void* d_out, int out_size, void* d_ws, size_t ws_size,
                              hipStream_t stream) {
    (void)in_sizes; (void)n_in; (void)out_size; (void)ws_size;
    constexpr int B = 32768, D_IN = 128, D_H = 1024;

    const float* x  = (const float*)d_in[0];
    const float* W1 = (const float*)d_in[1];
    const float* b1 = (const float*)d_in[2];
    const float* W2 = (const float*)d_in[3];
    const float* b2 = (const float*)d_in[4];
    float* out = (float*)d_out;

    char* ws = (char*)d_ws;
    int8_t*   xq    = (int8_t*)(ws + 0);
    int8_t*   w1q   = (int8_t*)(ws + 4194304);
    int8_t*   w2q   = (int8_t*)(ws + 4325376);
    int*      b1i   = (int*)   (ws + 5373952);
    int*      b2i   = (int*)   (ws + 5378048);
    float*    fs1   = (float*) (ws + 5382144);
    float*    fs2   = (float*) (ws + 5386240);
    unsigned* s1max = (unsigned*)(ws + 5390336);
    unsigned* hmax  = (unsigned*)(ws + 5390592);
    unsigned char* hq = (unsigned char*)(ws + 5390848);

    const dim3 gemm_grid(D_H / 64, B / 128);

    k_init<<<1, 32, 0, stream>>>(s1max, hmax);

    k_absmax<<<1024, 256, 0, stream>>>(x, B * D_IN, s1max);
    k_quant_x<<<(B * D_IN / 4 + 255) / 256, 256, 0, stream>>>(x, s1max, xq, B * D_IN / 4);
    k_quant_w<<<D_H, 256, 0, stream>>>(W1, b1, D_IN, s1max, 127.0f,
                                       w1q, b1i, fs1, nullptr, nullptr);
    gemm_i8<128, true, 0><<<gemm_grid, 256, 0, stream>>>(
        xq, w1q, b1i, fs1, nullptr, nullptr, nullptr, hmax, D_H);
    k_quant_w<<<D_H, 256, 0, stream>>>(W2, b2, D_H, hmax, 255.0f,
                                       w2q, b2i, fs2,
                                       out + (size_t)B * D_H + 1,
                                       out + (size_t)B * D_H);
    gemm_i8<128, true, 1><<<gemm_grid, 256, 0, stream>>>(
        xq, w1q, b1i, fs1, hmax, nullptr, hq, nullptr, D_H);
    gemm_i8<1024, false, 2><<<gemm_grid, 256, 0, stream>>>(
        (const int8_t*)hq, w2q, b2i, fs2, nullptr, out, nullptr, nullptr, D_H);
}
